// EdgeDecoder_21251498180835
// MI455X (gfx1250) — matrix operand order repaired and checked by reading, not compiled, not run on hardware
//
#include <hip/hip_runtime.h>
#include <hip/hip_bf16.h>

typedef __attribute__((ext_vector_type(16))) _Float16 v16h;
typedef __attribute__((ext_vector_type(8)))  float    v8f;

#define EMB 16
#define HID 64

// Build a 32x16 f16 B-fragment for V_WMMA_F32_16X16X32_F16 from a row-major
// f32 weight matrix W (row stride ldn). Fragment covers rows [row0,row0+32),
// cols [col0,col0+16). Layout: lanes 0-15 hold K=0..7,16..23 (halves 0..15),
// lanes 16-31 hold K=8..15,24..31.
__device__ __forceinline__ v16h load_w_frag(const float* __restrict__ W,
                                            int ldn, int row0, int col0,
                                            int lane) {
  const int n  = lane & 15;
  const int k0 = (lane >> 4) * 8;
  v16h b;
#pragma unroll
  for (int h = 0; h < 16; ++h)
    b[h] = (_Float16)W[(row0 + k0 + (h < 8 ? h : h + 8)) * ldn + (col0 + n)];
  return b;
}

__device__ __forceinline__ float sigmoidf_fast(float x) {
  return 1.0f / (1.0f + __expf(-x));
}

__global__ void __launch_bounds__(256) edge_mlp_wmma_kernel(
    const float* __restrict__ emb,   // [N_NODES, 16]
    const int*   __restrict__ eidx,  // [2, E]
    const float* __restrict__ W1,    // [32, 64]
    const float* __restrict__ b1,    // [64]
    const float* __restrict__ W2,    // [64, 32]
    const float* __restrict__ b2,    // [32]
    const float* __restrict__ W3,    // [32, 1]
    const float* __restrict__ b3,    // [1]
    float* __restrict__ out,         // [E]
    int E) {
  // Per-wave LDS staging of layer-1 output: 16 rows x 64 cols f32, stride 65
  // (stride 65 -> bank = (m + k) % 64, conflict-reduced).
  __shared__ float stage[8][16 * 65];

  const int lane = threadIdx.x & 31;
  const int wid  = threadIdx.x >> 5;
  float* __restrict__ h1 = stage[wid];

  const int n  = lane & 15;   // D-layout column of this lane
  const int hi = lane >> 4;   // upper half-wave flag

  // ---- weights: build WMMA B-fragments once, keep in VGPRs --------------
  v16h w1f[4];
#pragma unroll
  for (int t = 0; t < 4; ++t)
    w1f[t] = load_w_frag(W1, HID, 0, t * 16, lane);

  v16h w2f[2][2];  // [n_tile][k_block]
#pragma unroll
  for (int t = 0; t < 2; ++t)
#pragma unroll
    for (int kb = 0; kb < 2; ++kb)
      w2f[t][kb] = load_w_frag(W2, HID / 2, kb * 32, t * 16, lane);

  const float cb1[4] = { b1[n], b1[16 + n], b1[32 + n], b1[48 + n] };
  const float cb2[2] = { b2[n], b2[16 + n] };
  const float w3a = W3[n], w3b = W3[16 + n];
  const float bias3 = b3[0];

  const int* __restrict__ src = eidx;
  const int* __restrict__ dst = eidx + E;

  const int nTiles = (E + 15) >> 4;
  const int waveId = blockIdx.x * (blockDim.x >> 5) + wid;
  const int nWaves = gridDim.x * (blockDim.x >> 5);

  for (int tile = waveId; tile < nTiles; tile += nWaves) {
    const int eBase = tile << 4;
    const int e = min(eBase + n, E - 1);

    // Prefetch next tile's edge indices (global_prefetch_b8).
    const int nextBase = eBase + (nWaves << 4);
    if (nextBase < E) {
      __builtin_prefetch(&src[nextBase + n], 0, 1);
      __builtin_prefetch(&dst[nextBase + n], 0, 1);
    }

    // ---- gather: build A-fragment (16 edges x 32 feats, f16) ------------
    // Lane (m=n, hi) holds halves h=0..7 -> feats k = hi*8 + h   (src part)
    //                       halves h=8..15 -> feats k = 16+hi*8+(h-8) (dst)
    const int si = src[e];
    const int di = dst[e];
    const float4* sp = (const float4*)(emb + (size_t)si * EMB + hi * 8);
    const float4* dp = (const float4*)(emb + (size_t)di * EMB + hi * 8);
    const float4 s0 = sp[0], s1 = sp[1];
    const float4 d0 = dp[0], d1 = dp[1];

    v16h a;
    a[0] = (_Float16)s0.x;  a[1] = (_Float16)s0.y;
    a[2] = (_Float16)s0.z;  a[3] = (_Float16)s0.w;
    a[4] = (_Float16)s1.x;  a[5] = (_Float16)s1.y;
    a[6] = (_Float16)s1.z;  a[7] = (_Float16)s1.w;
    a[8] = (_Float16)d0.x;  a[9] = (_Float16)d0.y;
    a[10] = (_Float16)d0.z; a[11] = (_Float16)d0.w;
    a[12] = (_Float16)d1.x; a[13] = (_Float16)d1.y;
    a[14] = (_Float16)d1.z; a[15] = (_Float16)d1.w;

    // ---- layer 1: [16x32] @ [32x64] via 4 WMMAs, bias in C, fused ReLU --
#pragma unroll
    for (int t = 0; t < 4; ++t) {
      v8f c;
#pragma unroll
      for (int i = 0; i < 8; ++i) c[i] = cb1[t];
      c = __builtin_amdgcn_wmma_f32_16x16x32_f16(
              false, a, false, w1f[t], (short)0, c, false, false);
      // ReLU + stage D-tile to LDS as plain row-major 16x64 (stride 65).
#pragma unroll
      for (int r = 0; r < 8; ++r) {
        const int m = r + 8 * hi;             // D layout: VGPR r -> M
        h1[m * 65 + t * 16 + n] = fmaxf(c[r], 0.0f);
      }
    }
    asm volatile("s_wait_dscnt 0" ::: "memory");

    // ---- layer 2: [16x64] @ [64x32] via 2x2 chained WMMAs ---------------
    v8f acc2[2];
#pragma unroll
    for (int t = 0; t < 2; ++t)
#pragma unroll
      for (int i = 0; i < 8; ++i) acc2[t][i] = cb2[t];

#pragma unroll
    for (int kb = 0; kb < 2; ++kb) {
      v16h a2;  // A-layout rebuild from LDS (row m = n)
#pragma unroll
      for (int h = 0; h < 16; ++h) {
        const int k = kb * 32 + (h < 8 ? h : h + 8) + hi * 8;
        a2[h] = (_Float16)h1[n * 65 + k];
      }
#pragma unroll
      for (int t = 0; t < 2; ++t)
        acc2[t] = __builtin_amdgcn_wmma_f32_16x16x32_f16(
                      false, a2, false, w2f[t][kb], (short)0, acc2[t],
                      false, false);
    }

    // ---- layer 3: ReLU, dot with W3[32], cross-lane reduce, sigmoid -----
    float p[8];
#pragma unroll
    for (int r = 0; r < 8; ++r)
      p[r] = fmaxf(acc2[0][r], 0.0f) * w3a + fmaxf(acc2[1][r], 0.0f) * w3b;

#pragma unroll
    for (int off = 1; off <= 8; off <<= 1) {
#pragma unroll
      for (int r = 0; r < 8; ++r)
        p[r] += __shfl_xor(p[r], off, 32);
    }
    // Now every lane of each 16-lane half holds sums for m = r + 8*hi.
    if (n == 0) {
      float q[8];
#pragma unroll
      for (int r = 0; r < 8; ++r) q[r] = sigmoidf_fast(p[r] + bias3);
      const int mBase = eBase + hi * 8;
      if (mBase + 7 < E) {
        float4 o0 = { q[0], q[1], q[2], q[3] };
        float4 o1 = { q[4], q[5], q[6], q[7] };
        *(float4*)(out + mBase)     = o0;
        *(float4*)(out + mBase + 4) = o1;
      } else {
#pragma unroll
        for (int r = 0; r < 8; ++r)
          if (mBase + r < E) out[mBase + r] = q[r];
      }
    }
  }
}

extern "C" void kernel_launch(void* const* d_in, const int* in_sizes, int n_in,
                              void* d_out, int out_size, void* d_ws,
                              size_t ws_size, hipStream_t stream) {
  (void)n_in; (void)out_size; (void)d_ws; (void)ws_size;

  const float* emb = (const float*)d_in[0];
  const int*   eidx = (const int*)d_in[1];
  const float* W1 = (const float*)d_in[2];
  const float* b1 = (const float*)d_in[3];
  const float* W2 = (const float*)d_in[4];
  const float* b2 = (const float*)d_in[5];
  const float* W3 = (const float*)d_in[6];
  const float* b3 = (const float*)d_in[7];
  float* out = (float*)d_out;

  const int E = in_sizes[1] / 2;          // edge_index is [2, E]
  const int nTiles = (E + 15) / 16;       // one 16-edge tile per wave-iter
  const int wavesPerBlock = 256 / 32;     // 8 waves per block
  int blocks = (nTiles + wavesPerBlock - 1) / wavesPerBlock;
  if (blocks > 4096) blocks = 4096;       // grid-stride; amortize weight-frag setup

  edge_mlp_wmma_kernel<<<blocks, 256, 0, stream>>>(
      emb, eidx, W1, b1, W2, b2, W3, b3, out, E);

  // Tuple output #2: edge_index passthrough (bitwise int32 copy).
  hipMemcpyAsync(out + E, eidx, (size_t)2 * E * sizeof(int),
                 hipMemcpyDeviceToDevice, stream);
}
